// ResidualGatedGCN_28441273434190
// MI455X (gfx1250) — compile-verified, matrix operand order checked by reading, not run on hardware
//
#include <hip/hip_runtime.h>
#include <math.h>

typedef __attribute__((ext_vector_type(16))) __bf16 v16bf;
typedef __attribute__((ext_vector_type(8)))  float  v8f;

#define EMBED 128
#define EPT 4   // 16-edge tiles per wave in the edge kernel (B reuse factor)

__device__ __forceinline__ __bf16 to_bf16(float f) {
    unsigned u = __builtin_bit_cast(unsigned, f);
    u += 0x7FFFu + ((u >> 16) & 1u);            // round-to-nearest-even
    unsigned short h = (unsigned short)(u >> 16);
    return __builtin_bit_cast(__bf16, h);
}

// Load 16 contiguous f32 and convert to a bf16 fragment vector.
__device__ __forceinline__ v16bf load_cvt16(const float* __restrict__ p) {
    v16bf r;
    const float4* q = (const float4*)p;
#pragma unroll
    for (int i = 0; i < 4; ++i) {
        float4 f = q[i];
        r[4*i+0] = to_bf16(f.x);
        r[4*i+1] = to_bf16(f.y);
        r[4*i+2] = to_bf16(f.z);
        r[4*i+3] = to_bf16(f.w);
    }
    return r;
}

// ---------------------------------------------------------------------------
// Kernel 0: repack W [128,512] and We [128,128] (row-major, k-major) into
// column-major bf16: pack[n*128 + k].
// ---------------------------------------------------------------------------
__global__ __launch_bounds__(256) void pack_weights_kernel(
    const float* __restrict__ W, const float* __restrict__ We,
    __bf16* __restrict__ Wpack, __bf16* __restrict__ Wepack)
{
    int idx = blockIdx.x * blockDim.x + threadIdx.x;   // 0 .. 512*128-1
    if (idx < 512 * EMBED) {
        int n = idx / EMBED, k = idx % EMBED;
        Wpack[idx] = to_bf16(W[(size_t)k * 512 + n]);
    }
    if (idx < EMBED * EMBED) {
        int n = idx / EMBED, k = idx % EMBED;
        Wepack[idx] = to_bf16(We[(size_t)k * EMBED + n]);
    }
}

// ---------------------------------------------------------------------------
// Node GEMM body: proj = node_features @ W + bias; split h|Q|K|V.
// Block = 256 threads (8 waves), one 16-row M tile. Wave w: N-tiles [4w,4w+4).
// GUARD=false for full tiles (no per-element guards, imm-offset stores).
// ---------------------------------------------------------------------------
template<bool GUARD>
__device__ __forceinline__ void node_gemm_body(
    const float* __restrict__ node_features,
    const __bf16* __restrict__ Wpack,      // [512][128] bf16 col-major
    const float* __restrict__ W_bias,      // [512]
    float* __restrict__ out,               // h -> d_out [N,128]
    float* __restrict__ Qbuf,              // Q|K|V contiguous
    int n_nodes, int mtile, int tid)
{
    const int wave = tid >> 5;
    const int l    = tid & 31;
    const int ln   = l & 15;
    const int lh   = l >> 4;
    const size_t qkvStride = (size_t)n_nodes * EMBED;

    int arowi = mtile * 16 + ln;
    if (GUARD && arowi > n_nodes - 1) arowi = n_nodes - 1;
    const float* arow = node_features + (size_t)arowi * EMBED;

    v16bf a[4];
#pragma unroll
    for (int kt = 0; kt < 4; ++kt)
        a[kt] = load_cvt16(arow + kt * 32 + lh * 16);

#pragma unroll
    for (int t = 0; t < 4; ++t) {
        const int ntAbs = wave * 4 + t;                  // 0..31
        v8f acc = {};
#pragma unroll
        for (int kt = 0; kt < 4; ++kt) {
            v16bf b = *(const v16bf*)(Wpack + (size_t)(ntAbs * 16 + ln) * EMBED
                                             + kt * 32 + lh * 16);
            acc = __builtin_amdgcn_wmma_f32_16x16x32_bf16(
                      false, a[kt], false, b, (short)0, acc, false, false);
        }
        const float bias = W_bias[ntAbs * 16 + ln];
        const int   ncol = (ntAbs & 7) * 16 + ln;
        float* dst = (ntAbs < 8) ? out
                                 : Qbuf + (size_t)((ntAbs >> 3) - 1) * qkvStride;
        // One base pointer; rows r -> constant byte offsets (r*512B imm).
        // D layout: M = r + 8*lh.
        float* p = dst + (size_t)(mtile * 16 + 8 * lh) * EMBED + ncol;
#pragma unroll
        for (int r = 0; r < 8; ++r) {
            if (!GUARD || (mtile * 16 + 8 * lh + r) < n_nodes)
                p[(size_t)r * EMBED] = acc[r] + bias;
        }
    }
}

__global__ __launch_bounds__(256) void node_gemm_fast(
    const float* __restrict__ nf, const __bf16* __restrict__ Wp,
    const float* __restrict__ bias, float* __restrict__ out,
    float* __restrict__ Qbuf, int n_nodes)
{
    node_gemm_body<false>(nf, Wp, bias, out, Qbuf, n_nodes,
                          blockIdx.x, threadIdx.x);
}

__global__ __launch_bounds__(256) void node_gemm_tail(
    const float* __restrict__ nf, const __bf16* __restrict__ Wp,
    const float* __restrict__ bias, float* __restrict__ out,
    float* __restrict__ Qbuf, int n_nodes, int mtile0)
{
    node_gemm_body<true>(nf, Wp, bias, out, Qbuf, n_nodes,
                         mtile0 + blockIdx.x, threadIdx.x);
}

// ---------------------------------------------------------------------------
// Edge body: per 64-edge group (one wave, 4 M-tiles for 4x B reuse):
//   eproj = edge_features @ We + be   (WMMA, fused)
//   eta   = sigmoid(Q[recv] + K[send] + eproj)
//   atomicAdd(out[recv], eta * V[send])
// GUARD=false for full groups.
// ---------------------------------------------------------------------------
template<bool GUARD>
__device__ __forceinline__ void edge_body(
    const float* __restrict__ edge_features,
    const int* __restrict__ senders,
    const int* __restrict__ receivers,
    const __bf16* __restrict__ Wepack,     // [128][128] bf16 col-major
    const float* __restrict__ We_bias,     // [128]
    const float* __restrict__ Qbuf,
    const float* __restrict__ Kbuf,
    const float* __restrict__ Vbuf,
    float* __restrict__ out,
    int n_edges, int base, int l,
    int* sSend, int* sRecv)
{
    const int ln = l & 15;
    const int lh = l >> 4;

    for (int i = l; i < 16 * EPT; i += 32) {
        int e = base + i;
        if (GUARD && e > n_edges - 1) e = n_edges - 1;
        sSend[i] = senders[e];
        sRecv[i] = receivers[e];
    }
    __syncthreads();

    // A fragments for EPT edge tiles (kept resident; ~128 VGPRs).
    v16bf a[EPT][4];
#pragma unroll
    for (int t = 0; t < EPT; ++t) {
        int erow = base + t * 16 + ln;
        if (GUARD && erow > n_edges - 1) erow = n_edges - 1;
        const float* arow = edge_features + (size_t)erow * EMBED;
#pragma unroll
        for (int kt = 0; kt < 4; ++kt)
            a[t][kt] = load_cvt16(arow + kt * 32 + lh * 16);
    }

    // Row element-offsets per accumulator slot (D layout: edge = r + 8*lh).
    int srow[EPT][8], rrow[EPT][8];
#pragma unroll
    for (int t = 0; t < EPT; ++t)
#pragma unroll
        for (int r = 0; r < 8; ++r) {
            int idx = t * 16 + r + 8 * lh;
            srow[t][r] = sSend[idx] * EMBED;
            rrow[t][r] = sRecv[idx] * EMBED;
        }

#pragma unroll
    for (int nt = 0; nt < 8; ++nt) {
        // Load B fragments once, reuse across EPT tiles (4x L2 traffic cut).
        v16bf b[4];
#pragma unroll
        for (int kt = 0; kt < 4; ++kt)
            b[kt] = *(const v16bf*)(Wepack + (size_t)(nt * 16 + ln) * EMBED
                                            + kt * 32 + lh * 16);
        const int   n  = nt * 16 + ln;
        const float eb = We_bias[n];

#pragma unroll
        for (int t = 0; t < EPT; ++t) {
            v8f acc = {};
#pragma unroll
            for (int kt = 0; kt < 4; ++kt)
                acc = __builtin_amdgcn_wmma_f32_16x16x32_bf16(
                          false, a[t][kt], false, b[kt], (short)0, acc, false, false);

#pragma unroll
            for (int r = 0; r < 8; ++r) {
                if (!GUARD || (base + t * 16 + r + 8 * lh) < n_edges) {
                    int so = srow[t][r], ro = rrow[t][r];
                    float q   = Qbuf[ro + n];
                    float kk  = Kbuf[so + n] + acc[r] + eb;
                    float eta = 1.0f / (1.0f + __expf(-(q + kk)));
                    float msg = eta * Vbuf[so + n];
                    atomicAdd(&out[ro + n], msg);
                }
            }
        }
    }
}

__global__ __launch_bounds__(32) void edge_fast(
    const float* __restrict__ ef, const int* __restrict__ sn,
    const int* __restrict__ rc, const __bf16* __restrict__ Wep,
    const float* __restrict__ eb, const float* __restrict__ Qb,
    const float* __restrict__ Kb, const float* __restrict__ Vb,
    float* __restrict__ out, int n_edges)
{
    __shared__ int sSend[16 * EPT], sRecv[16 * EPT];
    edge_body<false>(ef, sn, rc, Wep, eb, Qb, Kb, Vb, out, n_edges,
                     blockIdx.x * (16 * EPT), threadIdx.x, sSend, sRecv);
}

__global__ __launch_bounds__(32) void edge_tail(
    const float* __restrict__ ef, const int* __restrict__ sn,
    const int* __restrict__ rc, const __bf16* __restrict__ Wep,
    const float* __restrict__ eb, const float* __restrict__ Qb,
    const float* __restrict__ Kb, const float* __restrict__ Vb,
    float* __restrict__ out, int n_edges, int group0)
{
    __shared__ int sSend[16 * EPT], sRecv[16 * EPT];
    edge_body<true>(ef, sn, rc, Wep, eb, Qb, Kb, Vb, out, n_edges,
                    (group0 + (int)blockIdx.x) * (16 * EPT), threadIdx.x,
                    sSend, sRecv);
}

// ---------------------------------------------------------------------------
extern "C" void kernel_launch(void* const* d_in, const int* in_sizes, int n_in,
                              void* d_out, int out_size, void* d_ws, size_t ws_size,
                              hipStream_t stream)
{
    const float* node_features = (const float*)d_in[0];
    const int*   senders       = (const int*)  d_in[1];
    const int*   receivers     = (const int*)  d_in[2];
    const float* edge_features = (const float*)d_in[3];
    const float* W_kernel      = (const float*)d_in[4];
    const float* W_bias        = (const float*)d_in[5];
    const float* We_kernel     = (const float*)d_in[6];
    const float* We_bias       = (const float*)d_in[7];
    float*       out           = (float*)d_out;

    const int n_nodes = in_sizes[0] / EMBED;
    const int n_edges = in_sizes[1];

    // Workspace layout: Q | K | V | Wpack(bf16) | Wepack(bf16)
    float*  Qbuf   = (float*)d_ws;
    float*  Kbuf   = Qbuf + (size_t)n_nodes * EMBED;
    float*  Vbuf   = Kbuf + (size_t)n_nodes * EMBED;
    __bf16* Wpack  = (__bf16*)(Vbuf + (size_t)n_nodes * EMBED);
    __bf16* Wepack = Wpack + (size_t)512 * EMBED;

    // 0) repack weights to bf16 column-major
    pack_weights_kernel<<<(512 * EMBED + 255) / 256, 256, 0, stream>>>(
        W_kernel, We_kernel, Wpack, Wepack);

    // 1) node projection GEMM (h -> d_out, Q/K/V -> ws)
    const int full_m = n_nodes / 16;
    if (full_m > 0)
        node_gemm_fast<<<full_m, 256, 0, stream>>>(
            node_features, Wpack, W_bias, out, Qbuf, n_nodes);
    if (n_nodes % 16)
        node_gemm_tail<<<1, 256, 0, stream>>>(
            node_features, Wpack, W_bias, out, Qbuf, n_nodes, full_m);

    // 2) fused edge GEMM + gather + gate + scatter-add
    const int full_e = n_edges / (16 * EPT);
    if (full_e > 0)
        edge_fast<<<full_e, 32, 0, stream>>>(
            edge_features, senders, receivers, Wepack, We_bias,
            Qbuf, Kbuf, Vbuf, out, n_edges);
    if (n_edges % (16 * EPT))
        edge_tail<<<1, 32, 0, stream>>>(
            edge_features, senders, receivers, Wepack, We_bias,
            Qbuf, Kbuf, Vbuf, out, n_edges, full_e);
}
